// LLMLayer_24378234372132
// MI455X (gfx1250) — matrix operand order repaired and checked by reading, not compiled, not run on hardware
//
#include <hip/hip_runtime.h>
#include <hip/hip_bf16.h>
#include <math.h>

#define L_SEQ 4096
#define DIM   1024
#define NH    8
#define DH    128
#define INTER 4096
#define EPS   1e-5f

typedef unsigned short u16;
typedef __attribute__((ext_vector_type(16))) __bf16 v16bf;
typedef __attribute__((ext_vector_type(8)))  float  v8f;

union FragU { v16bf v; uint4 q[2]; };

#if __has_builtin(__builtin_amdgcn_sched_group_barrier)
#define SGB(mask, cnt) __builtin_amdgcn_sched_group_barrier((mask), (cnt), 0)
#else
#define SGB(mask, cnt)
#endif
#define SG_VMEM_RD 0x020   // VMEM read group
#define SG_WMMA    0x008   // MFMA/WMMA group

__device__ __forceinline__ u16 f2bf(float f) {
    unsigned u = __float_as_uint(f);
    u += 0x7FFFu + ((u >> 16) & 1u);           // round-to-nearest-even
    return (u16)(u >> 16);
}

// A fragment (16x32 bf16, MxK): p = &A[(m0 + (lane&15))*lda + kb + ((lane>>4)<<4)]
__device__ __forceinline__ v16bf load_frag_a(const u16* p) {
    FragU f;
    f.q[0] = *(const uint4*)(p);        // halves 0..7  -> K chunk [k0, k0+8)
    f.q[1] = *(const uint4*)(p + 8);    // halves 8..15 -> K chunk [k0+8, k0+16)
    return f.v;
}

// B fragment (32x16 bf16, KxN) from W stored [N,K] row-major:
// p = &W[(n0 + (lane&15))*ldw + kb + ((lane>>4)<<4)]
__device__ __forceinline__ v16bf load_frag_b(const u16* p) {
    FragU f;
    f.q[0] = *(const uint4*)(p);        // halves 0..7  -> K [k0, k0+8)
    f.q[1] = *(const uint4*)(p + 8);    // halves 8..15 -> K [k0+8, k0+16)
    return f.v;
}

__device__ __forceinline__ v8f wmma_bf16(v16bf a, v16bf b, v8f c) {
    return __builtin_amdgcn_wmma_f32_16x16x32_bf16(
        /*neg_a=*/false, a, /*neg_b=*/false, b,
        /*c_mod=*/(short)0, c, /*reuse_a=*/false, /*reuse_b=*/false);
}

#define Z8 {0.f,0.f,0.f,0.f,0.f,0.f,0.f,0.f}

// ---------------------------------------------------------------- fp32->bf16
__global__ void __launch_bounds__(256) cvt_bf16_kernel(const float* __restrict__ src,
                                                       u16* __restrict__ dst, int n) {
    int i = blockIdx.x * 256 + threadIdx.x;
    if (i < n) dst[i] = f2bf(src[i]);
}

// ---------------------------------------------------------------- RMSNorm
__global__ void __launch_bounds__(256) rmsnorm_kernel(const float* __restrict__ x,
                                                      const float* __restrict__ w,
                                                      u16* __restrict__ out) {
    int row = blockIdx.x;
    const float* xr = x + (size_t)row * DIM;
    float s = 0.f;
    for (int i = threadIdx.x; i < DIM; i += 256) { float v = xr[i]; s += v * v; }
    for (int off = 16; off; off >>= 1) s += __shfl_xor(s, off, 32);
    __shared__ float red[8];
    if ((threadIdx.x & 31) == 0) red[threadIdx.x >> 5] = s;
    __syncthreads();
    float tot = 0.f;
    #pragma unroll
    for (int i = 0; i < 8; i++) tot += red[i];
    float rr = rsqrtf(tot / (float)DIM + EPS);
    for (int i = threadIdx.x; i < DIM; i += 256)
        out[(size_t)row * DIM + i] = f2bf(xr[i] * rr * w[i]);
}

// Scheduler pipeline directives for one fully-unrolled K range of NBLK blocks,
// 10 VMEM-reads + 4 WMMAs per block; shifts loads one block ahead of WMMAs.
__device__ __forceinline__ void gemm_sched(int j, int last) {
    if (j == 0)       { SGB(SG_VMEM_RD, 20); SGB(SG_WMMA, 4); }
    else if (j == last) {                    SGB(SG_WMMA, 4); }
    else              { SGB(SG_VMEM_RD, 10); SGB(SG_WMMA, 4); }
}

// ---------------------------------------------------------------- QKV GEMM
// qkv[L,3D] = h @ w_qkv^T + b;  scatter q->[H][L][dh], k->[H][L][dh], v->[H][dh][L]
__global__ void __launch_bounds__(128) gemm_qkv_kernel(const u16* __restrict__ A,
                                                       const u16* __restrict__ W,
                                                       const float* __restrict__ bias,
                                                       u16* __restrict__ qb,
                                                       u16* __restrict__ kb_,
                                                       u16* __restrict__ vtb) {
    const int lane = threadIdx.x & 31, wave = threadIdx.x >> 5;
    const int tid = blockIdx.x * 4 + wave;
    const int tm = tid % (L_SEQ / 16), tn = tid / (L_SEQ / 16);   // tn < 48
    const int m0 = tm * 16, n0 = tn * 64;
    const u16* pa = A + (size_t)(m0 + (lane & 15)) * DIM + ((lane >> 4) << 4);
    const u16* pb[4];
    #pragma unroll
    for (int t = 0; t < 4; t++)
        pb[t] = W + (size_t)(n0 + t * 16 + (lane & 15)) * DIM + ((lane >> 4) << 4);
    v8f acc[4] = {Z8, Z8, Z8, Z8};
    #pragma unroll
    for (int kb = 0; kb < DIM; kb += 32) {
        v16bf a = load_frag_a(pa + kb);
        #pragma unroll
        for (int t = 0; t < 4; t++) acc[t] = wmma_bf16(a, load_frag_b(pb[t] + kb), acc[t]);
        gemm_sched(kb, DIM - 32);
    }

    const int rbase = m0 + ((lane >> 4) << 3);
    #pragma unroll
    for (int t = 0; t < 4; t++) {
        int n = n0 + t * 16 + (lane & 15);
        float bz = bias[n];
        #pragma unroll
        for (int r = 0; r < 8; r++) {
            int m = rbase + r;
            u16 h = f2bf(acc[t][r] + bz);
            if (n < DIM)            { int hd = n >> 7, d = n & 127;            qb [((size_t)hd * L_SEQ + m) * DH + d] = h; }
            else if (n < 2 * DIM)   { int n2 = n - DIM; int hd = n2 >> 7, d = n2 & 127; kb_[((size_t)hd * L_SEQ + m) * DH + d] = h; }
            else                    { int n2 = n - 2 * DIM; int hd = n2 >> 7, d = n2 & 127; vtb[((size_t)hd * DH + d) * L_SEQ + m] = h; }
        }
    }
}

// ---------------------------------------------------------------- flash attention
// one wave = 16 query rows of one head; online softmax; V pre-transposed [H][dh][L]
__global__ void __launch_bounds__(128) attention_kernel(const u16* __restrict__ Q,
                                                        const u16* __restrict__ K,
                                                        const u16* __restrict__ VT,
                                                        u16* __restrict__ O) {
    const int lane = threadIdx.x & 31, wave = threadIdx.x >> 5;
    const int head = blockIdx.x;
    const int m0 = (blockIdx.y * 4 + wave) * 16;
    __shared__ u16 pstage[4][16 * 32];
    u16* ps = pstage[wave];

    const u16* Qh = Q  + (size_t)head * L_SEQ * DH;
    const u16* Kh = K  + (size_t)head * L_SEQ * DH;
    const u16* Vh = VT + (size_t)head * DH * L_SEQ;

    const u16* pa = Qh + (size_t)(m0 + (lane & 15)) * DH + ((lane >> 4) << 4);
    v16bf qf[4];
    #pragma unroll
    for (int i = 0; i < 4; i++) qf[i] = load_frag_a(pa + i * 32);

    v8f o[8] = {Z8, Z8, Z8, Z8, Z8, Z8, Z8, Z8};
    float rmax[8], rsum[8];
    #pragma unroll
    for (int r = 0; r < 8; r++) { rmax[r] = -INFINITY; rsum[r] = 0.f; }
    const float scl = 0.08838834764831845f;   // 1/sqrt(128)
    const int rowoff = (lane >> 4) << 3;

    for (int kb0 = 0; kb0 < L_SEQ; kb0 += 32) {
        // ---- burst-load all 8 K B-fragments, then 8 QK^T WMMAs
        const u16* pk0 = Kh + (size_t)(kb0 + (lane & 15)) * DH + ((lane >> 4) << 4);
        const u16* pk1 = pk0 + 16 * DH;
        v16bf kf[8];
        #pragma unroll
        for (int db = 0; db < 4; db++) {
            kf[db]     = load_frag_b(pk0 + db * 32);
            kf[4 + db] = load_frag_b(pk1 + db * 32);
        }
        v8f s0 = Z8, s1 = Z8;
        #pragma unroll
        for (int db = 0; db < 4; db++) {
            s0 = wmma_bf16(qf[db], kf[db], s0);
            s1 = wmma_bf16(qf[db], kf[4 + db], s1);
        }
        // ---- prefetch V fragments; their latency hides under the softmax VALU work
        const u16* pv = Vh + (size_t)(lane & 15) * L_SEQ + kb0 + ((lane >> 4) << 4);
        v16bf vf[8];
        #pragma unroll
        for (int dt = 0; dt < 8; dt++) vf[dt] = load_frag_b(pv + (size_t)dt * 16 * L_SEQ);

        // ---- online softmax (fp32)
        float a8[8];
        #pragma unroll
        for (int r = 0; r < 8; r++) {
            float v0 = s0[r] * scl, v1 = s1[r] * scl;
            float mx = fmaxf(v0, v1);
            #pragma unroll
            for (int off = 1; off < 16; off <<= 1) mx = fmaxf(mx, __shfl_xor(mx, off, 32));
            float nm = fmaxf(rmax[r], mx);
            float alpha = __expf(rmax[r] - nm);
            float p0 = __expf(v0 - nm), p1 = __expf(v1 - nm);
            float rs = p0 + p1;
            #pragma unroll
            for (int off = 1; off < 16; off <<= 1) rs += __shfl_xor(rs, off, 32);
            rsum[r] = rsum[r] * alpha + rs;
            rmax[r] = nm;
            a8[r] = alpha;
            int rowL = r + rowoff;
            ps[rowL * 32 + (lane & 15)]      = f2bf(p0);
            ps[rowL * 32 + 16 + (lane & 15)] = f2bf(p1);
        }
        #pragma unroll
        for (int dt = 0; dt < 8; dt++)
            #pragma unroll
            for (int r = 0; r < 8; r++) o[dt][r] *= a8[r];
        asm volatile("s_wait_dscnt 0" ::: "memory");   // P stores visible before re-layout load
        v16bf pf = load_frag_a(ps + (size_t)(lane & 15) * 32 + ((lane >> 4) << 4));
        #pragma unroll
        for (int dt = 0; dt < 8; dt++)
            o[dt] = wmma_bf16(pf, vf[dt], o[dt]);
        // schedule: K loads -> QK wmmas -> V loads -> PV wmmas (VALU/DS fill gaps)
        SGB(SG_VMEM_RD, 16);
        SGB(SG_WMMA, 8);
        SGB(SG_VMEM_RD, 16);
        SGB(SG_WMMA, 8);
    }
    #pragma unroll
    for (int dt = 0; dt < 8; dt++) {
        int col = head * DH + dt * 16 + (lane & 15);
        #pragma unroll
        for (int r = 0; r < 8; r++) {
            int m = m0 + rowoff + r;
            O[(size_t)m * DIM + col] = f2bf(o[dt][r] / rsum[r]);
        }
    }
}

// ---------------------------------------------------------------- proj + residual (fp32 out)
__global__ void __launch_bounds__(128) gemm_proj_kernel(const u16* __restrict__ A,
                                                        const u16* __restrict__ W,
                                                        const float* __restrict__ bias,
                                                        const float* __restrict__ resid,
                                                        float* __restrict__ X1) {
    const int lane = threadIdx.x & 31, wave = threadIdx.x >> 5;
    const int tid = blockIdx.x * 4 + wave;
    const int tm = tid % (L_SEQ / 16), tn = tid / (L_SEQ / 16);   // tn < 16
    const int m0 = tm * 16, n0 = tn * 64;
    const u16* pa = A + (size_t)(m0 + (lane & 15)) * DIM + ((lane >> 4) << 4);
    const u16* pb[4];
    #pragma unroll
    for (int t = 0; t < 4; t++)
        pb[t] = W + (size_t)(n0 + t * 16 + (lane & 15)) * DIM + ((lane >> 4) << 4);
    v8f acc[4] = {Z8, Z8, Z8, Z8};
    #pragma unroll
    for (int kb = 0; kb < DIM; kb += 32) {
        v16bf a = load_frag_a(pa + kb);
        #pragma unroll
        for (int t = 0; t < 4; t++) acc[t] = wmma_bf16(a, load_frag_b(pb[t] + kb), acc[t]);
        gemm_sched(kb, DIM - 32);
    }

    const int rbase = m0 + ((lane >> 4) << 3);
    #pragma unroll
    for (int t = 0; t < 4; t++) {
        int n = n0 + t * 16 + (lane & 15);
        float bz = bias[n];
        #pragma unroll
        for (int r = 0; r < 8; r++) {
            size_t idx = (size_t)(rbase + r) * DIM + n;
            X1[idx] = resid[idx] + acc[t][r] + bz;
        }
    }
}

// ---------------------------------------------------------------- gate+up (shared A stream, NT=2)
__global__ void __launch_bounds__(128) gemm_gateup_kernel(const u16* __restrict__ A,
                                                          const u16* __restrict__ Wg,
                                                          const u16* __restrict__ Wu,
                                                          u16* __restrict__ GU) {
    const int lane = threadIdx.x & 31, wave = threadIdx.x >> 5;
    const int tid = blockIdx.x * 4 + wave;
    const int tm = tid % (L_SEQ / 16), tn = tid / (L_SEQ / 16);   // tn < 128
    const int m0 = tm * 16, n0 = tn * 32;
    const u16* pa = A + (size_t)(m0 + (lane & 15)) * DIM + ((lane >> 4) << 4);
    const u16 *pg[2], *pu[2];
    #pragma unroll
    for (int t = 0; t < 2; t++) {
        size_t ro = (size_t)(n0 + t * 16 + (lane & 15)) * DIM + ((lane >> 4) << 4);
        pg[t] = Wg + ro; pu[t] = Wu + ro;
    }
    v8f ag[2] = {Z8, Z8}, au[2] = {Z8, Z8};
    #pragma unroll
    for (int kb = 0; kb < DIM; kb += 32) {
        v16bf a = load_frag_a(pa + kb);
        #pragma unroll
        for (int t = 0; t < 2; t++) {
            ag[t] = wmma_bf16(a, load_frag_b(pg[t] + kb), ag[t]);
            au[t] = wmma_bf16(a, load_frag_b(pu[t] + kb), au[t]);
        }
        gemm_sched(kb, DIM - 32);
    }

    const int rbase = m0 + ((lane >> 4) << 3);
    #pragma unroll
    for (int t = 0; t < 2; t++) {
        int n = n0 + t * 16 + (lane & 15);
        #pragma unroll
        for (int r = 0; r < 8; r++) {
            float g = ag[t][r];
            g = 0.5f * g * (1.f + erff(g * 0.7071067811865475f));   // exact GELU
            GU[(size_t)(rbase + r) * INTER + n] = f2bf(g * au[t][r]);
        }
    }
}

// ---------------------------------------------------------------- down + residual -> d_out
__global__ void __launch_bounds__(128) gemm_down_kernel(const u16* __restrict__ A,
                                                        const u16* __restrict__ W,
                                                        const float* __restrict__ resid,
                                                        float* __restrict__ out) {
    const int lane = threadIdx.x & 31, wave = threadIdx.x >> 5;
    const int tid = blockIdx.x * 4 + wave;
    const int tm = tid % (L_SEQ / 16), tn = tid / (L_SEQ / 16);   // tn < 16
    const int m0 = tm * 16, n0 = tn * 64;
    const u16* pa = A + (size_t)(m0 + (lane & 15)) * INTER + ((lane >> 4) << 4);
    const u16* pb[4];
    #pragma unroll
    for (int t = 0; t < 4; t++)
        pb[t] = W + (size_t)(n0 + t * 16 + (lane & 15)) * INTER + ((lane >> 4) << 4);
    v8f acc[4] = {Z8, Z8, Z8, Z8};
    // outer loop not unrolled; inner 16 blocks fully unrolled with pipelined sched groups
    for (int kb0 = 0; kb0 < INTER; kb0 += 512) {
        #pragma unroll
        for (int j = 0; j < 512; j += 32) {
            int kb = kb0 + j;
            v16bf a = load_frag_a(pa + kb);
            #pragma unroll
            for (int t = 0; t < 4; t++) acc[t] = wmma_bf16(a, load_frag_b(pb[t] + kb), acc[t]);
            gemm_sched(j, 512 - 32);
        }
    }

    const int rbase = m0 + ((lane >> 4) << 3);
    #pragma unroll
    for (int t = 0; t < 4; t++) {
        int n = n0 + t * 16 + (lane & 15);
        #pragma unroll
        for (int r = 0; r < 8; r++) {
            size_t idx = (size_t)(rbase + r) * DIM + n;
            out[idx] = resid[idx] + acc[t][r];
        }
    }
}

// ================================================================ launch
extern "C" void kernel_launch(void* const* d_in, const int* in_sizes, int n_in,
                              void* d_out, int out_size, void* d_ws, size_t ws_size,
                              hipStream_t stream) {
    const float* x     = (const float*)d_in[0];
    const float* wqkv  = (const float*)d_in[1];
    const float* bqkv  = (const float*)d_in[2];
    const float* wproj = (const float*)d_in[3];
    const float* bproj = (const float*)d_in[4];
    const float* ln1   = (const float*)d_in[5];
    const float* ln2   = (const float*)d_in[6];
    const float* wgate = (const float*)d_in[7];
    const float* wup   = (const float*)d_in[8];
    const float* wdown = (const float*)d_in[9];
    float* out = (float*)d_out;

    char* ws = (char*)d_ws;
    size_t off = 0;
    auto alloc = [&](size_t bytes) -> void* {
        void* p = ws + off;
        off += (bytes + 255) & ~(size_t)255;
        return p;
    };
    u16* wqkv_b  = (u16*)alloc((size_t)3 * DIM * DIM * 2);
    u16* wproj_b = (u16*)alloc((size_t)DIM * DIM * 2);
    u16* wgate_b = (u16*)alloc((size_t)INTER * DIM * 2);
    u16* wup_b   = (u16*)alloc((size_t)INTER * DIM * 2);
    u16* wdown_b = (u16*)alloc((size_t)DIM * INTER * 2);
    u16* h1      = (u16*)alloc((size_t)L_SEQ * DIM * 2);
    u16* qb      = (u16*)alloc((size_t)L_SEQ * DIM * 2);
    u16* kb      = (u16*)alloc((size_t)L_SEQ * DIM * 2);
    u16* vtb     = (u16*)alloc((size_t)L_SEQ * DIM * 2);
    u16* attnb   = (u16*)alloc((size_t)L_SEQ * DIM * 2);
    float* x1    = (float*)alloc((size_t)L_SEQ * DIM * 4);
    u16* h2      = (u16*)alloc((size_t)L_SEQ * DIM * 2);
    u16* gu      = (u16*)alloc((size_t)L_SEQ * INTER * 2);

    auto cvt = [&](const float* s, u16* d, int n) {
        cvt_bf16_kernel<<<(n + 255) / 256, 256, 0, stream>>>(s, d, n);
    };
    cvt(wqkv,  wqkv_b,  3 * DIM * DIM);
    cvt(wproj, wproj_b, DIM * DIM);
    cvt(wgate, wgate_b, INTER * DIM);
    cvt(wup,   wup_b,   INTER * DIM);
    cvt(wdown, wdown_b, DIM * INTER);

    rmsnorm_kernel<<<L_SEQ, 256, 0, stream>>>(x, ln1, h1);
    gemm_qkv_kernel<<<(L_SEQ / 16) * 48 / 4, 128, 0, stream>>>(h1, wqkv_b, bqkv, qb, kb, vtb);
    attention_kernel<<<dim3(NH, L_SEQ / 64), 128, 0, stream>>>(qb, kb, vtb, attnb);
    gemm_proj_kernel<<<(L_SEQ / 16) * 16 / 4, 128, 0, stream>>>(attnb, wproj_b, bproj, x, x1);
    rmsnorm_kernel<<<L_SEQ, 256, 0, stream>>>(x1, ln2, h2);
    gemm_gateup_kernel<<<(L_SEQ / 16) * 128 / 4, 128, 0, stream>>>(h2, wgate_b, wup_b, gu);
    gemm_down_kernel<<<(L_SEQ / 16) * 16 / 4, 128, 0, stream>>>(gu, wdown_b, x1, out);
}
